// VQVAEQuantizer_10986526343668
// MI455X (gfx1250) — compile-verified, matrix operand order checked by reading, not run on hardware
//
#include <hip/hip_runtime.h>
#include <hip/hip_bf16.h>
#include <stdint.h>

// ---------------------------------------------------------------------------
// VQ-VAE forward for MI455X (gfx1250), wave32, WMMA f32_16x16x32_f16.
// Compute-bound (~240 GF) -> matrix pipe in FP16 w/ FP32 accumulation.
// 4 independent accumulators per wave; ping-pong double-buffered fragments
// (two register sets alternating roles -> no rotation movs, <256 VGPRs).
// ---------------------------------------------------------------------------

typedef _Float16 half_t;
typedef __attribute__((ext_vector_type(16))) _Float16 v16h;
typedef __attribute__((ext_vector_type(4)))  _Float16 h4;
typedef __attribute__((ext_vector_type(8)))  float    v8f;

union FragU { v16h v; uint32_t u[8]; };

// Load a 16x32 fp16 fragment (A-layout per ISA 7.12.2; B uses the identical
// pattern with "row" = output column since B is stored [N,K] row-major).
__device__ __forceinline__ v16h load_frag(const half_t* tile, int ld) {
  const int lane = threadIdx.x & 31;
  const int r = lane & 15;      // matrix row (A) / column (B)
  const int h = lane >> 4;      // lane half selects K sub-block
  FragU f;
#pragma unroll
  for (int g = 0; g < 8; ++g) {
    const int k = (g < 4) ? (8 * h + 2 * g) : (16 + 8 * h + 2 * (g - 4));
    f.u[g] = *reinterpret_cast<const uint32_t*>(tile + (size_t)r * ld + k);
  }
  return f.v;
}

// ---------------------------------------------------------------------------
// Generic C[M,N] = A[M,K] @ B[N,K]^T + bias[N].  fp16 in, fp32 acc.
// Block: 256 thr (8 waves). Tile: 16 rows x 512 cols (16x64 per wave,
// 4 accumulators). A row-strip (16xK, contiguous) staged in LDS once.
// K-loop: ping-pong sets (chunk parity) -> 4 back-to-back WMMAs per chunk,
// refill loads overlap the other set's WMMAs.  K must be a multiple of 64.
// ---------------------------------------------------------------------------
template <int K>
__global__ void __launch_bounds__(256)
gemm_bt(const half_t* __restrict__ A, const half_t* __restrict__ B,
        const float* __restrict__ bias,
        float* __restrict__ Cf, half_t* __restrict__ Ch, int N) {
  __shared__ half_t sA[16 * K];
  const int row0 = blockIdx.y * 16;
  {
    const uint32_t* g = reinterpret_cast<const uint32_t*>(A + (size_t)row0 * K);
    uint32_t* s = reinterpret_cast<uint32_t*>(sA);
    for (int i = threadIdx.x; i < 8 * K; i += 256) s[i] = g[i];
  }
  __syncthreads();

  const int lane = threadIdx.x & 31;
  const int wv   = threadIdx.x >> 5;
  const int col0 = blockIdx.x * 512 + wv * 64;
  if (col0 >= N) return;

  const half_t* B0 = B + (size_t)col0 * K;
  v8f acc[4] = {{}, {}, {}, {}};

  // prologue: set A <- chunk 0, set B <- chunk 1
  v16h aA = load_frag(sA, K);
  v16h bA[4];
#pragma unroll
  for (int i = 0; i < 4; ++i) bA[i] = load_frag(B0 + (size_t)i * 16 * K, K);
  v16h aB = load_frag(sA + 32, K);
  v16h bB[4];
#pragma unroll
  for (int i = 0; i < 4; ++i) bB[i] = load_frag(B0 + (size_t)i * 16 * K + 32, K);

#pragma unroll 1
  for (int kk = 64; kk < K; kk += 64) {
    // consume set A (chunk kk-64), refill with chunk kk
#pragma unroll
    for (int i = 0; i < 4; ++i)
      acc[i] = __builtin_amdgcn_wmma_f32_16x16x32_f16(false, aA, false, bA[i],
                                                      (short)0, acc[i], false, false);
    aA = load_frag(sA + kk, K);
#pragma unroll
    for (int i = 0; i < 4; ++i) bA[i] = load_frag(B0 + (size_t)i * 16 * K + kk, K);
    // consume set B (chunk kk-32), refill with chunk kk+32
#pragma unroll
    for (int i = 0; i < 4; ++i)
      acc[i] = __builtin_amdgcn_wmma_f32_16x16x32_f16(false, aB, false, bB[i],
                                                      (short)0, acc[i], false, false);
    aB = load_frag(sA + kk + 32, K);
#pragma unroll
    for (int i = 0; i < 4; ++i) bB[i] = load_frag(B0 + (size_t)i * 16 * K + kk + 32, K);
  }
#pragma unroll
  for (int i = 0; i < 4; ++i)
    acc[i] = __builtin_amdgcn_wmma_f32_16x16x32_f16(false, aA, false, bA[i],
                                                    (short)0, acc[i], false, false);
#pragma unroll
  for (int i = 0; i < 4; ++i)
    acc[i] = __builtin_amdgcn_wmma_f32_16x16x32_f16(false, aB, false, bB[i],
                                                    (short)0, acc[i], false, false);

  const int rb = row0 + ((lane >> 4) << 3);
#pragma unroll
  for (int i = 0; i < 4; ++i) {
    const int col = col0 + i * 16 + (lane & 15);
    const float bb = bias[col];
#pragma unroll
    for (int g = 0; g < 8; ++g) {
      const float v = acc[i][g] + bb;
      const size_t off = (size_t)(rb + g) * N + col;
      if (Cf) Cf[off] = v;
      if (Ch) Ch[off] = (half_t)v;
    }
  }
}

// ---------------------------------------------------------------------------
// Fused codebook distance + argmin.  argmin ||z-c||^2 == argmax (z.c - .5|c|^2).
// Block handles 16 z_e rows (staged in LDS); each wave processes 4 consecutive
// codebook tiles per pass (4 accumulators, ping-pong pipelined), 16 passes
// cover 8192 codes; per-lane best -> shfl_xor reduce -> cross-wave LDS reduce.
// ---------------------------------------------------------------------------
__global__ void __launch_bounds__(256)
dist_argmax(const half_t* __restrict__ ZE, const half_t* __restrict__ CB,
            const float* __restrict__ cn, int* __restrict__ ids, int CODES) {
  constexpr int D = 1024;
  __shared__ half_t sZ[16 * D];
  __shared__ float sbv[8][16];
  __shared__ int   sbi[8][16];
  const int row0 = blockIdx.x * 16;
  {
    const uint32_t* g = reinterpret_cast<const uint32_t*>(ZE + (size_t)row0 * D);
    uint32_t* s = reinterpret_cast<uint32_t*>(sZ);
    for (int i = threadIdx.x; i < 8 * D; i += 256) s[i] = g[i];
  }
  __syncthreads();

  const int lane = threadIdx.x & 31;
  const int wv   = threadIdx.x >> 5;
  float bv[8]; int bi[8];
#pragma unroll
  for (int g = 0; g < 8; ++g) { bv[g] = -3.4e38f; bi[g] = 0x7fffffff; }

  const int npass = CODES / (16 * 32);  // 8 waves x 4 tiles per pass
#pragma unroll 1
  for (int pass = 0; pass < npass; ++pass) {
    const int t0 = pass * 32 + wv * 4;       // this wave's 4 consecutive tiles
    const half_t* B0 = CB + (size_t)t0 * 16 * D;
    if (pass + 1 < npass)
      __builtin_prefetch(CB + (size_t)(t0 + 32) * 16 * D, 0, 1);

    v8f acc[4] = {{}, {}, {}, {}};
    v16h aA = load_frag(sZ, D);
    v16h bA[4];
#pragma unroll
    for (int i = 0; i < 4; ++i) bA[i] = load_frag(B0 + (size_t)i * 16 * D, D);
    v16h aB = load_frag(sZ + 32, D);
    v16h bB[4];
#pragma unroll
    for (int i = 0; i < 4; ++i) bB[i] = load_frag(B0 + (size_t)i * 16 * D + 32, D);

#pragma unroll 1
    for (int kk = 64; kk < D; kk += 64) {
#pragma unroll
      for (int i = 0; i < 4; ++i)
        acc[i] = __builtin_amdgcn_wmma_f32_16x16x32_f16(false, aA, false, bA[i],
                                                        (short)0, acc[i], false, false);
      aA = load_frag(sZ + kk, D);
#pragma unroll
      for (int i = 0; i < 4; ++i) bA[i] = load_frag(B0 + (size_t)i * 16 * D + kk, D);
#pragma unroll
      for (int i = 0; i < 4; ++i)
        acc[i] = __builtin_amdgcn_wmma_f32_16x16x32_f16(false, aB, false, bB[i],
                                                        (short)0, acc[i], false, false);
      aB = load_frag(sZ + kk + 32, D);
#pragma unroll
      for (int i = 0; i < 4; ++i) bB[i] = load_frag(B0 + (size_t)i * 16 * D + kk + 32, D);
    }
#pragma unroll
    for (int i = 0; i < 4; ++i)
      acc[i] = __builtin_amdgcn_wmma_f32_16x16x32_f16(false, aA, false, bA[i],
                                                      (short)0, acc[i], false, false);
#pragma unroll
    for (int i = 0; i < 4; ++i)
      acc[i] = __builtin_amdgcn_wmma_f32_16x16x32_f16(false, aB, false, bB[i],
                                                      (short)0, acc[i], false, false);

#pragma unroll
    for (int i = 0; i < 4; ++i) {
      const int col = (t0 + i) * 16 + (lane & 15);
      const float c2 = cn[col];
#pragma unroll
      for (int g = 0; g < 8; ++g) {
        const float s = acc[i][g] - c2;
        if (s > bv[g] || (s == bv[g] && col < bi[g])) { bv[g] = s; bi[g] = col; }
      }
    }
  }
  // reduce across the 16 lanes of each half (masks 1..8 stay within a half)
#pragma unroll
  for (int m = 8; m >= 1; m >>= 1) {
#pragma unroll
    for (int g = 0; g < 8; ++g) {
      const float ov = __shfl_xor(bv[g], m, 32);
      const int   oi = __shfl_xor(bi[g], m, 32);
      if (ov > bv[g] || (ov == bv[g] && oi < bi[g])) { bv[g] = ov; bi[g] = oi; }
    }
  }
  if ((lane & 15) == 0) {
    const int hf = lane >> 4;  // lane 0 -> rows 0..7, lane 16 -> rows 8..15
#pragma unroll
    for (int g = 0; g < 8; ++g) { sbv[wv][hf * 8 + g] = bv[g]; sbi[wv][hf * 8 + g] = bi[g]; }
  }
  __syncthreads();
  if (threadIdx.x < 16) {
    float best = sbv[0][threadIdx.x]; int idx = sbi[0][threadIdx.x];
#pragma unroll
    for (int w = 1; w < 8; ++w) {
      const float v = sbv[w][threadIdx.x]; const int i2 = sbi[w][threadIdx.x];
      if (v > best || (v == best && i2 < idx)) { best = v; idx = i2; }
    }
    ids[row0 + threadIdx.x] = idx;
  }
}

// ---------------------------------------------------------------------------
// Helpers: fp32->fp16 convert, codebook half-norms, gather, loss reductions.
// ---------------------------------------------------------------------------
__global__ void cvt_f2h(const float* __restrict__ s, half_t* __restrict__ d, long n) {
  const long stride = (long)gridDim.x * blockDim.x;
  for (long j = (long)blockIdx.x * blockDim.x + threadIdx.x; j * 4 < n; j += stride) {
    const float4 v = reinterpret_cast<const float4*>(s)[j];
    h4 o; o.x = (half_t)v.x; o.y = (half_t)v.y; o.z = (half_t)v.z; o.w = (half_t)v.w;
    reinterpret_cast<h4*>(d)[j] = o;
  }
}

__global__ void __launch_bounds__(256)
cb_norms(const float* __restrict__ cb, float* __restrict__ cn) {
  __shared__ float red[256];
  const int row = blockIdx.x;
  float s = 0.f;
  for (int c = threadIdx.x; c < 1024; c += 256) {
    const float v = cb[(size_t)row * 1024 + c]; s += v * v;
  }
  red[threadIdx.x] = s; __syncthreads();
  for (int off = 128; off > 0; off >>= 1) {
    if (threadIdx.x < off) red[threadIdx.x] += red[threadIdx.x + off];
    __syncthreads();
  }
  if (!threadIdx.x) cn[row] = 0.5f * red[0];
}

__global__ void gather_zq(const float* __restrict__ cb, const int* __restrict__ ids,
                          float* __restrict__ zf, half_t* __restrict__ zh) {
  const int row = blockIdx.x;
  const int id  = ids[row];
  for (int c = threadIdx.x; c < 1024; c += blockDim.x) {
    const float v = cb[(size_t)id * 1024 + c];
    zf[(size_t)row * 1024 + c] = v;
    zh[(size_t)row * 1024 + c] = (half_t)v;
  }
}

__global__ void __launch_bounds__(256)
sqdiff_part(const float* __restrict__ a, const float* __restrict__ b,
            float* __restrict__ part, long n) {
  __shared__ float red[256];
  const long base = (long)blockIdx.x * 2048 + threadIdx.x;
  float s = 0.f;
#pragma unroll
  for (int i = 0; i < 8; ++i) {
    const long idx = base + (long)i * 256;
    if (idx < n) { const float d = a[idx] - b[idx]; s += d * d; }
  }
  red[threadIdx.x] = s; __syncthreads();
  for (int off = 128; off > 0; off >>= 1) {
    if (threadIdx.x < off) red[threadIdx.x] += red[threadIdx.x + off];
    __syncthreads();
  }
  if (!threadIdx.x) part[blockIdx.x] = red[0];
}

__global__ void __launch_bounds__(256)
finalize(const float* __restrict__ pr, int nr, const float* __restrict__ pe, int ne,
         float* __restrict__ out) {
  __shared__ float red[256];
  float s = 0.f;
  for (int i = threadIdx.x; i < nr; i += 256) s += pr[i];
  red[threadIdx.x] = s; __syncthreads();
  for (int off = 128; off > 0; off >>= 1) {
    if (threadIdx.x < off) red[threadIdx.x] += red[threadIdx.x + off];
    __syncthreads();
  }
  const float sum_r = red[0]; __syncthreads();
  s = 0.f;
  for (int i = threadIdx.x; i < ne; i += 256) s += pe[i];
  red[threadIdx.x] = s; __syncthreads();
  for (int off = 128; off > 0; off >>= 1) {
    if (threadIdx.x < off) red[threadIdx.x] += red[threadIdx.x + off];
    __syncthreads();
  }
  if (!threadIdx.x) {
    const float sum_e = red[0];
    const float recon = sum_r / (float)((long)8192 * 2048);
    const float emb   = sum_e / (float)((long)8192 * 1024);
    out[0] = recon + emb + 0.25f * emb;  // vq_loss (commit fwd value == emb)
    out[1] = recon;
    out[2] = emb;
    out[3] = emb;
  }
}

// ---------------------------------------------------------------------------
extern "C" void kernel_launch(void* const* d_in, const int* in_sizes, int n_in,
                              void* d_out, int out_size, void* d_ws, size_t ws_size,
                              hipStream_t stream) {
  (void)in_sizes; (void)n_in; (void)out_size; (void)ws_size;
  const float* roi   = (const float*)d_in[0];
  const float* W_in  = (const float*)d_in[1];
  const float* b_in  = (const float*)d_in[2];
  const float* W_enc = (const float*)d_in[3];
  const float* b_enc = (const float*)d_in[4];
  const float* cb    = (const float*)d_in[5];
  const float* W_dec = (const float*)d_in[6];
  const float* b_dec = (const float*)d_in[7];
  const float* W_out = (const float*)d_in[8];
  const float* b_out = (const float*)d_in[9];

  constexpr long N = 8192, IN = 2048, D = 1024, KC = 8192;

  float* out_f   = (float*)d_out;
  int*   ids     = (int*)d_out;            // [N] int32 bit-pattern
  float* zq_f    = out_f + N;              // [N, D]
  float* recon   = zq_f + N * D;           // [N, IN]
  float* loss    = recon + N * IN;         // 4 scalars

  char* w = (char*)d_ws;
  auto carve = [&](size_t bytes) { char* p = w; w += (bytes + 255) & ~(size_t)255; return p; };
  half_t* roi_h  = (half_t*)carve(N * IN * 2);
  half_t* Win_h  = (half_t*)carve(D * IN * 2);
  half_t* Wenc_h = (half_t*)carve(D * D * 2);
  half_t* cb_h   = (half_t*)carve(KC * D * 2);
  half_t* Wdec_h = (half_t*)carve(D * D * 2);
  half_t* Wout_h = (half_t*)carve(IN * D * 2);
  half_t* h1_h   = (half_t*)carve(N * D * 2);
  float*  ze_f   = (float*)carve(N * D * 4);
  half_t* ze_h   = (half_t*)carve(N * D * 2);
  half_t* zq_h   = (half_t*)carve(N * D * 2);
  half_t* h3_h   = (half_t*)carve(N * D * 2);
  float*  cn     = (float*)carve(KC * 4);
  float*  part_r = (float*)carve(8192 * 4);
  float*  part_e = (float*)carve(4096 * 4);

  // fp32 -> fp16 operand staging
  cvt_f2h<<<1024, 256, 0, stream>>>(roi,   roi_h,  N * IN);
  cvt_f2h<<<512,  256, 0, stream>>>(W_in,  Win_h,  D * IN);
  cvt_f2h<<<256,  256, 0, stream>>>(W_enc, Wenc_h, D * D);
  cvt_f2h<<<1024, 256, 0, stream>>>(cb,    cb_h,   KC * D);
  cvt_f2h<<<256,  256, 0, stream>>>(W_dec, Wdec_h, D * D);
  cvt_f2h<<<512,  256, 0, stream>>>(W_out, Wout_h, IN * D);
  cb_norms<<<KC, 256, 0, stream>>>(cb, cn);

  // encoder  (col tile = 512 per workgroup)
  gemm_bt<2048><<<dim3(D / 512, N / 16), 256, 0, stream>>>(roi_h, Win_h, b_in, nullptr, h1_h, (int)D);
  gemm_bt<1024><<<dim3(D / 512, N / 16), 256, 0, stream>>>(h1_h, Wenc_h, b_enc, ze_f, ze_h, (int)D);
  // nearest codebook entry (fused distance + argmin)
  dist_argmax<<<N / 16, 256, 0, stream>>>(ze_h, cb_h, cn, ids, (int)KC);
  gather_zq<<<N, 256, 0, stream>>>(cb, ids, zq_f, zq_h);
  // decoder
  gemm_bt<1024><<<dim3(D / 512, N / 16), 256, 0, stream>>>(zq_h, Wdec_h, b_dec, nullptr, h3_h, (int)D);
  gemm_bt<1024><<<dim3(IN / 512, N / 16), 256, 0, stream>>>(h3_h, Wout_h, b_out, recon, nullptr, (int)IN);

  // losses (deterministic two-stage reductions)
  sqdiff_part<<<8192, 256, 0, stream>>>(recon, roi, part_r, N * IN);
  sqdiff_part<<<4096, 256, 0, stream>>>(zq_f, ze_f, part_e, N * D);
  finalize<<<1, 256, 0, stream>>>(part_r, 8192, part_e, 4096, loss);
}
